// SAGEFeaturePropagation_13778255085922
// MI455X (gfx1250) — matrix operand order repaired and checked by reading, not compiled, not run on hardware
//
#include <hip/hip_runtime.h>
#include <hip/hip_bf16.h>

typedef __attribute__((ext_vector_type(16))) __bf16 v16bf;
typedef __attribute__((ext_vector_type(8)))  __bf16 v8bf;
typedef __attribute__((ext_vector_type(4)))  __bf16 v4bf;
typedef __attribute__((ext_vector_type(8)))  float  v8f;

#define D 128

// ---------------- zero workspace (agg + deg) ----------------
__global__ void zero_ws_kernel(float* __restrict__ p, long long n) {
    long long i = (long long)blockIdx.x * blockDim.x + threadIdx.x;
    if (i < n) p[i] = 0.0f;
}

// ---------------- edge scatter-add (mean aggregation numerator + degree) ----
// one wave32 per edge; lane l handles features [4l, 4l+4)
__global__ void __launch_bounds__(256) scatter_kernel(
        const float* __restrict__ x,
        const int*   __restrict__ ei,   // [2, E] flattened: row=ei[e], col=ei[E+e]
        float* __restrict__ agg,
        float* __restrict__ deg,
        int E_) {
    const int wave = threadIdx.x >> 5;
    const int lane = threadIdx.x & 31;
    const int e = blockIdx.x * 8 + wave;
    if (e >= E_) return;
    const int row = ei[e];
    const int col = ei[E_ + e];
    const float4 xv = *(const float4*)(x + (long long)col * D + lane * 4);
    float* dst = agg + (long long)row * D + lane * 4;
    atomicAdd(dst + 0, xv.x);
    atomicAdd(dst + 1, xv.y);
    atomicAdd(dst + 2, xv.z);
    atomicAdd(dst + 3, xv.w);
    if (lane == 0) atomicAdd(deg + row, 1.0f);
}

// ---------------- fold the two matmuls: WcT[m][i] = sum_j lin_w[j][i]*weight[j][m]
//                  bc[m] = sum_j lin_b[j]*weight[j][m]
__global__ void fuse_weights_kernel(
        const float* __restrict__ lin_w,   // [D_OUT, D_IN]
        const float* __restrict__ lin_b,   // [D_OUT]
        const float* __restrict__ weight,  // [D_OUT, D_OUT]
        __bf16* __restrict__ WcT,          // [m][i] (pre-transposed for B-fragment loads)
        float*  __restrict__ bc) {
    const int t = blockIdx.x * blockDim.x + threadIdx.x;   // 0..16383
    const int i = t >> 7;
    const int m = t & 127;
    float s = 0.0f;
    for (int j = 0; j < D; ++j) s += lin_w[j * D + i] * weight[j * D + m];
    WcT[m * D + i] = (__bf16)s;
    if (t < D) {
        float sb = 0.0f;
        for (int j = 0; j < D; ++j) sb += lin_b[j] * weight[j * D + t];
        bc[t] = sb;
    }
}

// ---------------- normalize by degree, cast to bf16 ----------------
__global__ void normalize_kernel(
        const float* __restrict__ agg,
        const float* __restrict__ deg,
        __bf16* __restrict__ h, int Nn) {
    const int g = blockIdx.x * blockDim.x + threadIdx.x;   // N*32 groups of 4 feats
    if (g >= Nn * 32) return;
    const int n = g >> 5;
    const int f = (g & 31) * 4;
    float d = deg[n];
    d = (d < 1.0f) ? 1.0f : d;
    const float inv = 1.0f / d;
    const float4 a = *(const float4*)(agg + (long long)n * D + f);
    v4bf o;
    o[0] = (__bf16)(a.x * inv);
    o[1] = (__bf16)(a.y * inv);
    o[2] = (__bf16)(a.z * inv);
    o[3] = (__bf16)(a.w * inv);
    *(v4bf*)(h + (long long)n * D + f) = o;
}

// ---------------- fused GEMM: out[n,m] = sum_k h[n,k]*WcT[m,k] + bc[m] --------
// block = 8 waves; block owns 16-row stripe, wave w owns 16-col tile w.
// K=128 -> 4 x v_wmma_f32_16x16x32_bf16, fp32 accumulation.
__global__ void __launch_bounds__(256) gemm_kernel(
        const __bf16* __restrict__ h,     // [N, 128] bf16 row-major
        const __bf16* __restrict__ WcT,   // [128(m), 128(k)] bf16 row-major
        const float*  __restrict__ bc,    // [128]
        float* __restrict__ out, int Nn) {
    const int lane = threadIdx.x & 31;
    const int wave = threadIdx.x >> 5;        // column tile 0..7
    const int r0   = blockIdx.x * 16;         // row tile base
    const int hl   = lane & 15;
    const int hi   = lane >> 4;

    // A fragment (ISA 16-bit A 16x32): lane row = hl; VGPR0-3 hold K=kb..kb+7,
    // VGPR4-7 hold K=16+kb..16+kb+7 with kb = hi*8  -> two contiguous 16B loads.
    const __bf16* aptr = h + (long long)(r0 + hl) * D + hi * 8;
    // B fragment (32x16 bf16 B): lane col = hl; same K order as A (kb..kb+7, 16+kb..16+kb+7).
    const int n = wave * 16 + hl;
    const __bf16* bptr = WcT + (long long)n * D + hi * 8;

    v8f acc = {};
#pragma unroll
    for (int k = 0; k < 4; ++k) {
        v16bf a, b;
        *((v8bf*)&a)     = *(const v8bf*)(aptr + k * 32);
        *((v8bf*)&a + 1) = *(const v8bf*)(aptr + k * 32 + 16);
        *((v8bf*)&b)     = *(const v8bf*)(bptr + k * 32);
        *((v8bf*)&b + 1) = *(const v8bf*)(bptr + k * 32 + 16);
        acc = __builtin_amdgcn_wmma_f32_16x16x32_bf16(
                  false, a, false, b, (short)0, acc, false, false);
    }

    // C/D layout: VGPR r -> row (r + 8*hi), col hl within tile.
    const float bias = bc[n];
    float* optr = out + (long long)r0 * D + n;
#pragma unroll
    for (int r = 0; r < 8; ++r) {
        optr[(long long)(r + hi * 8) * D] = acc[r] + bias;
    }
}

extern "C" void kernel_launch(void* const* d_in, const int* in_sizes, int n_in,
                              void* d_out, int out_size, void* d_ws, size_t ws_size,
                              hipStream_t stream) {
    const float* x      = (const float*)d_in[0];
    const int*   ei     = (const int*)  d_in[1];
    const float* lin_w  = (const float*)d_in[2];
    const float* lin_b  = (const float*)d_in[3];
    const float* weight = (const float*)d_in[4];

    const int Nn = in_sizes[0] / D;   // 100000
    const int E_ = in_sizes[1] / 2;   // 800000

    // workspace layout
    float*  agg = (float*)d_ws;                      // N*128 f32
    float*  deg = agg + (size_t)Nn * D;              // N f32
    __bf16* h   = (__bf16*)(deg + Nn);               // N*128 bf16
    __bf16* WcT = h + (size_t)Nn * D;                // 128*128 bf16
    float*  bc  = (float*)(WcT + D * D);             // 128 f32
    float*  out = (float*)d_out;

    const long long zn = (long long)Nn * D + Nn;
    zero_ws_kernel<<<(int)((zn + 255) / 256), 256, 0, stream>>>(agg, zn);
    scatter_kernel<<<(E_ + 7) / 8, 256, 0, stream>>>(x, ei, agg, deg, E_);
    fuse_weights_kernel<<<(D * D) / 256, 256, 0, stream>>>(lin_w, lin_b, weight, WcT, bc);
    normalize_kernel<<<(Nn * 32 + 255) / 256, 256, 0, stream>>>(agg, deg, h, Nn);
    gemm_kernel<<<Nn / 16, 256, 0, stream>>>(h, WcT, bc, out, Nn);
}
